// LoRAFastLinear_32040456028925
// MI455X (gfx1250) — matrix operand order repaired and checked by reading, not compiled, not run on hardware
//
#include <hip/hip_runtime.h>

// ---------------------------------------------------------------------------
// LoRA fused linear for MI455X (gfx1250, wave32).
//   out = X @ W^T + bias + sum_l scale_l * (X @ down_l^T) @ up_l^T
// Split-bf16 (hi/lo) emulated-fp32 GEMM on v_wmma_f32_16x16x32_bf16:
//   x*w ~= xh*wh + xh*wl + xl*wh   (f32 accumulation inside WMMA)
// Pipelining: ping-pong LDS buffers; bf16 operand tiles staged with
// GLOBAL_LOAD_ASYNC_TO_LDS_B128 (ASYNCcnt), X f32 tiles prefetched to VGPRs
// during compute and split->ds_store before the single per-step barrier.
// ---------------------------------------------------------------------------

typedef __bf16 bf16_t;
typedef __attribute__((ext_vector_type(8)))  __bf16 v8bf;
typedef __attribute__((ext_vector_type(16))) __bf16 v16bf;
typedef __attribute__((ext_vector_type(8)))  float  v8f;

#define IN_DIM   2048
#define OUT_DIM  2048
#define M_TOT    16384   // B*S
#define R_RANK   32
#define KLORA    128     // L*R

#define TM  128          // block tile M
#define TN  128          // block tile N
#define TK  32           // K chunk (one WMMA K)
#define LDK 40           // padded LDS row stride (80B: 16B aligned, conflict-free)

struct alignas(16) LdsTiles {
  bf16_t Ah[TM * LDK];
  bf16_t Al[TM * LDK];
  bf16_t Bh[TN * LDK];
  bf16_t Bl[TN * LDK];
};

__device__ __forceinline__ v16bf cat8(v8bf a, v8bf b) {
  return __builtin_shufflevector(a, b, 0, 1, 2, 3, 4, 5, 6, 7, 8, 9, 10, 11, 12, 13, 14, 15);
}

__device__ __forceinline__ void split1(float v, bf16_t& h, bf16_t& l) {
  bf16_t hh = (bf16_t)v;
  h = hh;
  l = (bf16_t)(v - (float)hh);
}

// ---- CDNA5 async global->LDS copy (ASYNCcnt) ------------------------------
__device__ __forceinline__ void async_b128(const bf16_t* g, const bf16_t* s) {
  asm volatile("global_load_async_to_lds_b128 %0, %1, off"
               :: "v"((unsigned)(size_t)s), "v"((unsigned long long)(size_t)g)
               : "memory");
}
__device__ __forceinline__ void wait_async0() {
  asm volatile("s_wait_asynccnt 0x0" ::: "memory");
}

// Stage one 128 x 32 bf16 tile (row-major, leading dim ld) into padded LDS.
__device__ __forceinline__ void stage_async(const bf16_t* __restrict__ g, int ld,
                                            int rowbase, int colbase,
                                            bf16_t* __restrict__ S, int t) {
  const int row = t >> 1, c0 = (t & 1) * 16;
  const bf16_t* gp = g + (size_t)(rowbase + row) * ld + colbase + c0;
  const bf16_t* sp = S + row * LDK + c0;
  async_b128(gp, sp);
  async_b128(gp + 8, sp + 8);
}

// ---- X f32 register prefetch + split->LDS ---------------------------------
struct XReg { float4 f[4]; };

__device__ __forceinline__ XReg load_x(const float* __restrict__ x,
                                       int rowbase, int colbase, int t) {
  const int row = t >> 1, c0 = (t & 1) * 16;
  const float4* p = (const float4*)(x + (size_t)(rowbase + row) * IN_DIM + colbase + c0);
  XReg r;
  r.f[0] = p[0]; r.f[1] = p[1]; r.f[2] = p[2]; r.f[3] = p[3];
  return r;
}

__device__ __forceinline__ void store_x_split(const XReg& r, bf16_t* __restrict__ Sh,
                                              bf16_t* __restrict__ Sl, int t) {
  const int row = t >> 1, c0 = (t & 1) * 16;
  float vals[16] = {r.f[0].x, r.f[0].y, r.f[0].z, r.f[0].w,
                    r.f[1].x, r.f[1].y, r.f[1].z, r.f[1].w,
                    r.f[2].x, r.f[2].y, r.f[2].z, r.f[2].w,
                    r.f[3].x, r.f[3].y, r.f[3].z, r.f[3].w};
  v8bf h0, h1, l0, l1;
#pragma unroll
  for (int i = 0; i < 8; ++i) {
    bf16_t hh, ll;
    split1(vals[i], hh, ll);     h0[i] = hh; l0[i] = ll;
    split1(vals[i + 8], hh, ll); h1[i] = hh; l1[i] = ll;
  }
  *(v8bf*)(Sh + row * LDK + c0)     = h0;
  *(v8bf*)(Sh + row * LDK + c0 + 8) = h1;
  *(v8bf*)(Sl + row * LDK + c0)     = l0;
  *(v8bf*)(Sl + row * LDK + c0 + 8) = l1;
}

// ---- One K-chunk of split-bf16 WMMA. Wave = 32(M) x 64(N) sub-tile. -------
// Waves arranged 4(M) x 2(N) within the 128x128 block tile.
__device__ __forceinline__ void wmma_tiles(const LdsTiles& sm, int wave, int lane,
                                           v8f acc[2][4]) {
  const int half = lane >> 4, lm = lane & 15;
  const int wm = (wave & 3) * 32, wn = (wave >> 2) * 64;
  v16bf ah[2], al[2];
#pragma unroll
  for (int mt = 0; mt < 2; ++mt) {
    const bf16_t* ph = sm.Ah + (wm + mt * 16 + lm) * LDK;
    const bf16_t* pl = sm.Al + (wm + mt * 16 + lm) * LDK;
    ah[mt] = cat8(*(const v8bf*)(ph + 8 * half), *(const v8bf*)(ph + 16 + 8 * half));
    al[mt] = cat8(*(const v8bf*)(pl + 8 * half), *(const v8bf*)(pl + 16 + 8 * half));
  }
#pragma unroll
  for (int nt = 0; nt < 4; ++nt) {
    const bf16_t* pbh = sm.Bh + (wn + nt * 16 + lm) * LDK + 8 * half;
    const bf16_t* pbl = sm.Bl + (wn + nt * 16 + lm) * LDK + 8 * half;
    v16bf bh = cat8(*(const v8bf*)pbh, *(const v8bf*)(pbh + 16));
    v16bf bl = cat8(*(const v8bf*)pbl, *(const v8bf*)(pbl + 16));
#pragma unroll
    for (int mt = 0; mt < 2; ++mt) {
      acc[mt][nt] = __builtin_amdgcn_wmma_f32_16x16x32_bf16(false, ah[mt], false, bh, (short)0, acc[mt][nt], false, false);
      acc[mt][nt] = __builtin_amdgcn_wmma_f32_16x16x32_bf16(false, ah[mt], false, bl, (short)0, acc[mt][nt], false, false);
      acc[mt][nt] = __builtin_amdgcn_wmma_f32_16x16x32_bf16(false, al[mt], false, bh, (short)0, acc[mt][nt], false, false);
    }
  }
}

// ---------------------------------------------------------------------------
// Prep: split f32 array into bf16 hi/lo.
__global__ void split_kernel(const float* __restrict__ src, bf16_t* __restrict__ h,
                             bf16_t* __restrict__ l, int n) {
  int i = blockIdx.x * blockDim.x + threadIdx.x;
  if (i < n) {
    bf16_t hh, ll;
    split1(src[i], hh, ll);
    h[i] = hh;
    l[i] = ll;
  }
}

// Prep: U2[n][l*32+r] = scales[l] * ups[l][n][r], split hi/lo. n-major, ld=128.
__global__ void u2_kernel(const float* __restrict__ ups, const float* __restrict__ scales,
                          bf16_t* __restrict__ uh, bf16_t* __restrict__ ul, int n) {
  int i = blockIdx.x * blockDim.x + threadIdx.x;
  if (i < n) {
    int nrow = i >> 7;
    int c = i & 127;
    int li = c >> 5;
    int r = c & 31;
    float v = scales[li] * ups[((size_t)li * OUT_DIM + nrow) * R_RANK + r];
    bf16_t hh, ll;
    split1(v, hh, ll);
    uh[i] = hh;
    ul[i] = ll;
  }
}

// ---------------------------------------------------------------------------
// tmp[m][l*32+r] = X @ downs^T  (K=2048, N=128), stored as bf16 hi/lo.
__global__ __launch_bounds__(256) void lora_tmp_kernel(const float* __restrict__ x,
                                                       const bf16_t* __restrict__ Dhi,
                                                       const bf16_t* __restrict__ Dlo,
                                                       bf16_t* __restrict__ Thi,
                                                       bf16_t* __restrict__ Tlo) {
  __shared__ LdsTiles sm[2];
  const int t = threadIdx.x;
  const int wave = t >> 5, lane = t & 31;
  const int mbase = blockIdx.x * TM;
  v8f acc[2][4];
#pragma unroll
  for (int i = 0; i < 2; ++i)
#pragma unroll
    for (int j = 0; j < 4; ++j) acc[i][j] = v8f{};

  // Prologue: stage tile 0.
  stage_async(Dhi, IN_DIM, 0, 0, sm[0].Bh, t);
  stage_async(Dlo, IN_DIM, 0, 0, sm[0].Bl, t);
  XReg xr = load_x(x, mbase, 0, t);
  store_x_split(xr, sm[0].Ah, sm[0].Al, t);
  wait_async0();
  __syncthreads();

  int buf = 0;
  for (int kk = 0; kk < IN_DIM; kk += TK) {
    const int nb = buf ^ 1;
    const int kn = kk + TK;
    const bool more = kn < IN_DIM;
    if (more) {
      stage_async(Dhi, IN_DIM, 0, kn, sm[nb].Bh, t);
      stage_async(Dlo, IN_DIM, 0, kn, sm[nb].Bl, t);
      xr = load_x(x, mbase, kn, t);
    }
    wmma_tiles(sm[buf], wave, lane, acc);
    if (more) store_x_split(xr, sm[nb].Ah, sm[nb].Al, t);
    wait_async0();
    __syncthreads();
    buf = nb;
  }

  const int half = lane >> 4, lm = lane & 15;
  const int wm = (wave & 3) * 32, wn = (wave >> 2) * 64;
#pragma unroll
  for (int mt = 0; mt < 2; ++mt) {
    const int mrow = mbase + wm + mt * 16 + half * 8;
#pragma unroll
    for (int nt = 0; nt < 4; ++nt) {
      const int n = wn + nt * 16 + lm;
#pragma unroll
      for (int j = 0; j < 8; ++j) {
        bf16_t hh, ll;
        split1(acc[mt][nt][j], hh, ll);
        Thi[(size_t)(mrow + j) * KLORA + n] = hh;
        Tlo[(size_t)(mrow + j) * KLORA + n] = ll;
      }
    }
  }
}

// ---------------------------------------------------------------------------
// Main: out = X @ W^T + bias + tmp @ U2   (K = 2048, then K' = 128)
__global__ __launch_bounds__(256) void lora_main_kernel(
    const float* __restrict__ x, const float* __restrict__ bias,
    const bf16_t* __restrict__ Whi, const bf16_t* __restrict__ Wlo,
    const bf16_t* __restrict__ Thi, const bf16_t* __restrict__ Tlo,
    const bf16_t* __restrict__ Uhi, const bf16_t* __restrict__ Ulo,
    float* __restrict__ out) {
  __shared__ LdsTiles sm[2];
  const int t = threadIdx.x;
  const int wave = t >> 5, lane = t & 31;
  const int mbase = blockIdx.x * TM;
  const int nbase = blockIdx.y * TN;
  v8f acc[2][4];
#pragma unroll
  for (int i = 0; i < 2; ++i)
#pragma unroll
    for (int j = 0; j < 4; ++j) acc[i][j] = v8f{};

  // ---- Phase 1: base GEMM over K = IN_DIM, pipelined ----
  stage_async(Whi, IN_DIM, nbase, 0, sm[0].Bh, t);
  stage_async(Wlo, IN_DIM, nbase, 0, sm[0].Bl, t);
  XReg xr = load_x(x, mbase, 0, t);
  store_x_split(xr, sm[0].Ah, sm[0].Al, t);
  wait_async0();
  __syncthreads();

  int buf = 0;
  for (int kk = 0; kk < IN_DIM; kk += TK) {
    const int nb = buf ^ 1;
    const int kn = kk + TK;
    const bool more = kn < IN_DIM;
    if (more) {
      stage_async(Whi, IN_DIM, nbase, kn, sm[nb].Bh, t);
      stage_async(Wlo, IN_DIM, nbase, kn, sm[nb].Bl, t);
      xr = load_x(x, mbase, kn, t);
    }
    wmma_tiles(sm[buf], wave, lane, acc);
    if (more) store_x_split(xr, sm[nb].Ah, sm[nb].Al, t);
    wait_async0();
    __syncthreads();
    buf = nb;
  }

  // ---- Phase 2: LoRA delta over K' = 128 (all operands bf16: full async) ----
  for (int kk = 0; kk < KLORA; kk += TK) {
    stage_async(Thi, KLORA, mbase, kk, sm[0].Ah, t);
    stage_async(Tlo, KLORA, mbase, kk, sm[0].Al, t);
    stage_async(Uhi, KLORA, nbase, kk, sm[0].Bh, t);
    stage_async(Ulo, KLORA, nbase, kk, sm[0].Bl, t);
    wait_async0();
    __syncthreads();
    wmma_tiles(sm[0], wave, lane, acc);
    __syncthreads();
  }

  // ---- Epilogue: bias + store. C elem j -> m = j + 8*half, n = lane&15. ----
  const int half = lane >> 4, lm = lane & 15;
  const int wm = (wave & 3) * 32, wn = (wave >> 2) * 64;
#pragma unroll
  for (int mt = 0; mt < 2; ++mt) {
    const int mrow = mbase + wm + mt * 16 + half * 8;
#pragma unroll
    for (int nt = 0; nt < 4; ++nt) {
      const int n = nbase + wn + nt * 16 + lm;
      const float bv = bias[n];
      const size_t base = (size_t)mrow * OUT_DIM + n;
#pragma unroll
      for (int j = 0; j < 8; ++j)
        out[base + (size_t)j * OUT_DIM] = acc[mt][nt][j] + bv;
    }
  }
}

// ---------------------------------------------------------------------------
extern "C" void kernel_launch(void* const* d_in, const int* in_sizes, int n_in,
                              void* d_out, int out_size, void* d_ws, size_t ws_size,
                              hipStream_t stream) {
  const float* x      = (const float*)d_in[0];  // [B,S,IN] = [16384, 2048]
  const float* weight = (const float*)d_in[1];  // [OUT, IN]
  const float* bias   = (const float*)d_in[2];  // [OUT]
  const float* downs  = (const float*)d_in[3];  // [L,R,IN] = [128, 2048] flat
  const float* ups    = (const float*)d_in[4];  // [L,OUT,R]
  const float* scales = (const float*)d_in[5];  // [L]
  float* out = (float*)d_out;

  // Workspace carve-up (~27.3 MB total, all 16B-aligned chunks).
  char* ws = (char*)d_ws;
  bf16_t* Whi = (bf16_t*)ws; ws += (size_t)OUT_DIM * IN_DIM * sizeof(bf16_t);
  bf16_t* Wlo = (bf16_t*)ws; ws += (size_t)OUT_DIM * IN_DIM * sizeof(bf16_t);
  bf16_t* Dhi = (bf16_t*)ws; ws += (size_t)KLORA * IN_DIM * sizeof(bf16_t);
  bf16_t* Dlo = (bf16_t*)ws; ws += (size_t)KLORA * IN_DIM * sizeof(bf16_t);
  bf16_t* Uhi = (bf16_t*)ws; ws += (size_t)OUT_DIM * KLORA * sizeof(bf16_t);
  bf16_t* Ulo = (bf16_t*)ws; ws += (size_t)OUT_DIM * KLORA * sizeof(bf16_t);
  bf16_t* Thi = (bf16_t*)ws; ws += (size_t)M_TOT * KLORA * sizeof(bf16_t);
  bf16_t* Tlo = (bf16_t*)ws; ws += (size_t)M_TOT * KLORA * sizeof(bf16_t);

  const int nW = OUT_DIM * IN_DIM;
  split_kernel<<<(nW + 255) / 256, 256, 0, stream>>>(weight, Whi, Wlo, nW);
  const int nD = KLORA * IN_DIM;
  split_kernel<<<(nD + 255) / 256, 256, 0, stream>>>(downs, Dhi, Dlo, nD);
  const int nU = OUT_DIM * KLORA;
  u2_kernel<<<(nU + 255) / 256, 256, 0, stream>>>(ups, scales, Uhi, Ulo, nU);

  lora_tmp_kernel<<<dim3(M_TOT / TM), 256, 0, stream>>>(x, Dhi, Dlo, Thi, Tlo);
  lora_main_kernel<<<dim3(M_TOT / TM, OUT_DIM / TN), 256, 0, stream>>>(
      x, bias, Whi, Wlo, Thi, Tlo, Uhi, Ulo, out);
}
